// ImageEncoderViT_91070486545048
// MI455X (gfx1250) — matrix operand order repaired and checked by reading, not compiled, not run on hardware
//
#include <hip/hip_runtime.h>
#include <math.h>

// ---------------------------------------------------------------------------
// ImageEncoderViT (SAM-style) forward for gfx1250 (MI455X).
// All dense math goes through a WMMA f16 (f32-accum) batched GEMM.
// Block tile 128x64, 8 waves (wave32), each wave owns a 32x32 C tile
// (4 accumulators -> 4 v_wmma per k-step). Double-buffered LDS with a
// single barrier per k-step + global_prefetch of the k+64 tile.
// ---------------------------------------------------------------------------

typedef __attribute__((ext_vector_type(16))) _Float16 v16h;
typedef __attribute__((ext_vector_type(8)))  float    v8f;

#define GEMM_GELU 1

__global__ __launch_bounds__(256) void gemm_wmma_f16(
    const float* __restrict__ A, long long aOffO, long long aOffI, int lda,
    const float* __restrict__ B, long long bOffO, long long bOffI, int ldb,
    float* __restrict__ C, long long cOffO, long long cOffI, int ldc,
    const float* __restrict__ bias, const float* __restrict__ resid,
    int M, int N, int K, int inner, float alpha, int flags, int transB)
{
  const int z  = blockIdx.z;
  const int zo = z / inner;
  const int zi = z - zo * inner;
  A += zo * aOffO + zi * aOffI;
  B += zo * bOffO + zi * bOffI;
  C += zo * cOffO + zi * cOffI;

  __shared__ _Float16 As [2][128][40];   // [buf][m][k], row = 80B (16B aligned)
  __shared__ _Float16 BsT[2][ 64][40];   // [buf][n][k], transposed B tile

  const int tid  = threadIdx.x;
  const int lane = tid & 31;
  const int wave = tid >> 5;
  const int wm   = wave & 3;          // 4 waves down  (32 rows each)
  const int wn   = wave >> 2;         // 2 waves across (32 cols each)
  const int rowBase = blockIdx.y * 128;
  const int colBase = blockIdx.x * 64;

  const int m  = lane & 15;
  const int hi = lane >> 4;

  const v8f zero8 = {0.f,0.f,0.f,0.f,0.f,0.f,0.f,0.f};
  v8f acc00 = zero8, acc01 = zero8, acc10 = zero8, acc11 = zero8;

  // staging maps
  const int arow = tid >> 1;          // 0..127
  const int acol = (tid & 1) << 4;    // 0 or 16
  const int btn  = tid >> 2;          // 0..63  (transB: C-column)
  const int btk  = (tid & 3) << 3;    // 0,8,16,24
  const int bk   = tid >> 3;          // 0..31  (normal: K row)
  const int bn   = (tid & 7) << 3;    // 0..56

  float aR[16];
  float bR[8];

  // ---- load one k-tile (A:16 floats, B:8 floats per thread) into registers
  auto loadTiles = [&](int k0) {
    const bool kFull = (k0 + 32 <= K);
    // A tile
    if ((rowBase + 128 <= M) && kFull) {           // uniform fast path
      const float* ap = A + (long long)(rowBase + arow) * lda + k0 + acol;
      const float4 v0 = ((const float4*)ap)[0];
      const float4 v1 = ((const float4*)ap)[1];
      const float4 v2 = ((const float4*)ap)[2];
      const float4 v3 = ((const float4*)ap)[3];
      aR[ 0]=v0.x; aR[ 1]=v0.y; aR[ 2]=v0.z; aR[ 3]=v0.w;
      aR[ 4]=v1.x; aR[ 5]=v1.y; aR[ 6]=v1.z; aR[ 7]=v1.w;
      aR[ 8]=v2.x; aR[ 9]=v2.y; aR[10]=v2.z; aR[11]=v2.w;
      aR[12]=v3.x; aR[13]=v3.y; aR[14]=v3.z; aR[15]=v3.w;
      if (k0 + 64 <= K) __builtin_prefetch(ap + 32, 0, 3);   // global_prefetch
    } else {                                        // clamped, branchless edge
      const int gm = rowBase + arow;
      const long long rA = (long long)(gm < M ? gm : M - 1) * lda;
      #pragma unroll
      for (int j = 0; j < 16; ++j) {
        const int gk = k0 + acol + j;
        const float v = A[rA + (gk < K ? gk : K - 1)];
        aR[j] = (gm < M && gk < K) ? v : 0.f;
      }
    }
    // B tile
    const bool fullB = (colBase + 64 <= N) && kFull;   // uniform
    if (transB) {                                   // B is (N,K): contiguous in K
      const int gn = colBase + btn;
      if (fullB) {
        const float* bp = B + (long long)gn * ldb + k0 + btk;
        const float4 v0 = ((const float4*)bp)[0];
        const float4 v1 = ((const float4*)bp)[1];
        bR[0]=v0.x; bR[1]=v0.y; bR[2]=v0.z; bR[3]=v0.w;
        bR[4]=v1.x; bR[5]=v1.y; bR[6]=v1.z; bR[7]=v1.w;
        if (k0 + 64 <= K) __builtin_prefetch(bp + 32, 0, 3);
      } else {
        const long long rB = (long long)(gn < N ? gn : N - 1) * ldb;
        #pragma unroll
        for (int j = 0; j < 8; ++j) {
          const int gk = k0 + btk + j;
          const float v = B[rB + (gk < K ? gk : K - 1)];
          bR[j] = (gn < N && gk < K) ? v : 0.f;
        }
      }
    } else {                                        // B is (K,N): contiguous in N
      const int gk = k0 + bk;
      if (fullB) {
        const float* bp = B + (long long)gk * ldb + colBase + bn;
        const float4 v0 = ((const float4*)bp)[0];
        const float4 v1 = ((const float4*)bp)[1];
        bR[0]=v0.x; bR[1]=v0.y; bR[2]=v0.z; bR[3]=v0.w;
        bR[4]=v1.x; bR[5]=v1.y; bR[6]=v1.z; bR[7]=v1.w;
        if (k0 + 64 <= K) __builtin_prefetch(bp + 32LL * ldb, 0, 3);
      } else {
        const long long rB = (long long)(gk < K ? gk : K - 1) * ldb;
        #pragma unroll
        for (int j = 0; j < 8; ++j) {
          const int gn = colBase + bn + j;
          const float v = B[rB + (gn < N ? gn : N - 1)];
          bR[j] = (gn < N && gk < K) ? v : 0.f;
        }
      }
    }
  };

  // ---- convert registers -> f16 and store into LDS buffer `buf`
  auto storeTiles = [&](int buf) {
    _Float16* dA = &As[buf][arow][acol];
    #pragma unroll
    for (int j = 0; j < 16; ++j) dA[j] = (_Float16)aR[j];
    if (transB) {
      _Float16* dB = &BsT[buf][btn][btk];
      #pragma unroll
      for (int j = 0; j < 8; ++j) dB[j] = (_Float16)bR[j];
    } else {
      #pragma unroll
      for (int j = 0; j < 8; ++j) BsT[buf][bn + j][bk] = (_Float16)bR[j];
    }
  };

  // ---- prologue: stage tile 0
  loadTiles(0);
  storeTiles(0);
  __syncthreads();

  int cur = 0;
  for (int k0 = 0; k0 < K; k0 += 32) {
    const bool hasNext = (k0 + 32 < K);
    if (hasNext) loadTiles(k0 + 32);       // global loads overlap WMMA below

    // -------- fragments (contiguous k = 16*hi + s, wave32) --------
    v16h a0, a1, b0, b1;
    const _Float16* ar0 = &As[cur][wm * 32 + m][0];
    const _Float16* ar1 = &As[cur][wm * 32 + 16 + m][0];
    #pragma unroll
    for (int p = 0; p < 8; ++p) {          // A: K = p + 16*hi (+8 for upper run)
      a0[p]     = ar0[hi * 16 + p];
      a0[p + 8] = ar0[hi * 16 + 8 + p];
      a1[p]     = ar1[hi * 16 + p];
      a1[p + 8] = ar1[hi * 16 + 8 + p];
    }
    const _Float16* br0 = &BsT[cur][wn * 32 + m][hi * 16];    // B: K = p + 16*hi
    const _Float16* br1 = &BsT[cur][wn * 32 + 16 + m][hi * 16];
    #pragma unroll
    for (int p = 0; p < 16; ++p) { b0[p] = br0[p]; b1[p] = br1[p]; }

    acc00 = __builtin_amdgcn_wmma_f32_16x16x32_f16(false, a0, false, b0, (short)0, acc00, false, false);
    acc01 = __builtin_amdgcn_wmma_f32_16x16x32_f16(false, a0, false, b1, (short)0, acc01, false, false);
    acc10 = __builtin_amdgcn_wmma_f32_16x16x32_f16(false, a1, false, b0, (short)0, acc10, false, false);
    acc11 = __builtin_amdgcn_wmma_f32_16x16x32_f16(false, a1, false, b1, (short)0, acc11, false, false);

    if (hasNext) storeTiles(cur ^ 1);      // write next tile into other buffer
    __syncthreads();
    cur ^= 1;
  }

  // -------- epilogue: C layout = VGPR r -> row r + 8*hi, col = lane&15 --------
  #pragma unroll
  for (int s = 0; s < 2; ++s) {
    #pragma unroll
    for (int r = 0; r < 8; ++r) {
      const int gm = rowBase + wm * 32 + s * 16 + r + hi * 8;
      if (gm >= M) continue;
      #pragma unroll
      for (int t = 0; t < 2; ++t) {
        const int gn = colBase + wn * 32 + t * 16 + m;
        if (gn >= N) continue;
        float v = (s == 0 ? (t == 0 ? acc00[r] : acc01[r])
                          : (t == 0 ? acc10[r] : acc11[r])) * alpha;
        if (bias)  v += bias[gn];
        if (flags & GEMM_GELU) v = 0.5f * v * (1.f + erff(v * 0.70710678118654752f));
        if (resid) v += resid[(long long)gm * ldc + gn];
        C[(long long)gm * ldc + gn] = v;
      }
    }
  }
}

// ===========================================================================
// Elementwise / reduction helper kernels
// ===========================================================================
__global__ __launch_bounds__(256) void im2col_patch(const float* __restrict__ x,
                                                    float* __restrict__ out)
{
  const long long idx = (long long)blockIdx.x * 256 + threadIdx.x;
  if (idx >= 4096LL * 768) return;
  const int col = (int)(idx % 768);
  const long long p = idx / 768;
  const int pw = (int)(p % 64), ph = (int)(p / 64);
  const int c = col >> 8, rem = col & 255, py = rem >> 4, px = rem & 15;
  out[idx] = x[((long long)c * 1024 + ph * 16 + py) * 1024 + pw * 16 + px];
}

__global__ __launch_bounds__(256) void add_inplace(float* __restrict__ a,
                                                   const float* __restrict__ b,
                                                   long long n)
{
  const long long idx = (long long)blockIdx.x * 256 + threadIdx.x;
  if (idx < n) a[idx] += b[idx];
}

__global__ __launch_bounds__(256) void layer_norm_row(
    const float* __restrict__ in, const float* __restrict__ w,
    const float* __restrict__ b, float* __restrict__ out, int C, float eps)
{
  const int row = blockIdx.x, tid = threadIdx.x;
  const float* x = in + (long long)row * C;
  float* y = out + (long long)row * C;
  float s = 0.f, s2 = 0.f;
  for (int c = tid; c < C; c += 256) { const float v = x[c]; s += v; s2 += v * v; }
  __shared__ float sh1[256], sh2[256];
  sh1[tid] = s; sh2[tid] = s2; __syncthreads();
  for (int o = 128; o > 0; o >>= 1) {
    if (tid < o) { sh1[tid] += sh1[tid + o]; sh2[tid] += sh2[tid + o]; }
    __syncthreads();
  }
  const float mean = sh1[0] / C;
  const float var  = sh2[0] / C - mean * mean;
  const float inv  = rsqrtf(var + eps);
  for (int c = tid; c < C; c += 256) y[c] = (x[c] - mean) * inv * w[c] + b[c];
}

// window partition (64,64,768) -> (25,14,14,768) with zero pad (70x70)
__global__ __launch_bounds__(256) void win_part(const float* __restrict__ in,
                                                float* __restrict__ out)
{
  const long long idx = (long long)blockIdx.x * 256 + threadIdx.x;
  if (idx >= 4900LL * 768) return;
  const int d = (int)(idx % 768);
  long long t = idx / 768;
  const int j = (int)(t % 14); t /= 14;
  const int i = (int)(t % 14); t /= 14;
  const int wv = (int)t;                      // 0..24
  const int h = (wv / 5) * 14 + i, w = (wv % 5) * 14 + j;
  out[idx] = (h < 64 && w < 64) ? in[((long long)h * 64 + w) * 768 + d] : 0.f;
}

// x += unpartition(win)
__global__ __launch_bounds__(256) void win_unpart_add(float* __restrict__ x,
                                                      const float* __restrict__ win)
{
  const long long idx = (long long)blockIdx.x * 256 + threadIdx.x;
  if (idx >= 4096LL * 768) return;
  const int d = (int)(idx % 768);
  const long long t = idx / 768;
  const int h = (int)(t / 64), w = (int)(t % 64);
  const int wv = (h / 14) * 5 + (w / 14);
  x[idx] += win[(((long long)wv * 14 + (h % 14)) * 14 + (w % 14)) * 768 + d];
}

// rel_h / rel_w: out[bh, hq, w, kq] = sum_c q[bh, hq*W+w, c] * table[coord - kq + S - 1, c]
__global__ __launch_bounds__(256) void rel_pos_kernel(
    const float* __restrict__ qkv, const float* __restrict__ table,
    float* __restrict__ out, long long total, int Ntok, int H, int W, int S,
    int axis, int heads, int ldq)
{
  const long long idx = (long long)blockIdx.x * 256 + threadIdx.x;
  if (idx >= total) return;
  const int kq = (int)(idx % S);
  long long r = idx / S;
  const int w = (int)(r % W); r /= W;
  const int hq = (int)(r % H); r /= H;
  const long long bh = r;
  const long long winIdx = bh / heads;
  const int head = (int)(bh % heads);
  const int tok = hq * W + w;
  const float* qp = qkv + (winIdx * Ntok + tok) * (long long)ldq + head * 64;
  const int d = (axis == 0 ? hq : w) - kq + S - 1;
  const float* tr = table + (long long)d * 64;
  float acc = 0.f;
  #pragma unroll
  for (int c = 0; c < 64; ++c) acc += qp[c] * tr[c];
  out[idx] = acc;  // layout (bh*Ntok + tok)*S + kq
}

// in-place: scores[row,:] = softmax(scores[row,:] + relh[kh] + relw[kw])
__global__ __launch_bounds__(256) void softmax_bias(
    float* __restrict__ scores, const float* __restrict__ relh,
    const float* __restrict__ relw, int Ntok, int W)
{
  const long long row = blockIdx.x;
  const int q = (int)(row % Ntok);
  const long long bh = row / Ntok;
  const int H = Ntok / W;
  float* s = scores + row * (long long)Ntok;
  const float* rh = relh + (bh * (long long)Ntok + q) * H;
  const float* rw = relw + (bh * (long long)Ntok + q) * W;
  const int tid = threadIdx.x;
  __shared__ float red[256];

  float mx = -3.0e38f;
  for (int k = tid; k < Ntok; k += 256) {
    const int kh = k / W, kw = k % W;
    const float v = s[k] + rh[kh] + rw[kw];
    s[k] = v;
    mx = fmaxf(mx, v);
  }
  red[tid] = mx; __syncthreads();
  for (int o = 128; o > 0; o >>= 1) {
    if (tid < o) red[tid] = fmaxf(red[tid], red[tid + o]);
    __syncthreads();
  }
  mx = red[0]; __syncthreads();

  float sum = 0.f;
  for (int k = tid; k < Ntok; k += 256) { const float e = expf(s[k] - mx); s[k] = e; sum += e; }
  red[tid] = sum; __syncthreads();
  for (int o = 128; o > 0; o >>= 1) {
    if (tid < o) red[tid] += red[tid + o];
    __syncthreads();
  }
  const float inv = 1.f / red[0];
  for (int k = tid; k < Ntok; k += 256) s[k] *= inv;
}

// channel layernorm over 256 channels; optional transposed (NCHW) output
__global__ __launch_bounds__(256) void ln2d_c256(
    const float* __restrict__ in, const float* __restrict__ w,
    const float* __restrict__ b, float* __restrict__ out, int npos, float eps,
    int transposed_out)
{
  const int pos = blockIdx.x, c = threadIdx.x;
  const float v = in[(long long)pos * 256 + c];
  __shared__ float sh1[256], sh2[256];
  sh1[c] = v; sh2[c] = v * v; __syncthreads();
  for (int o = 128; o > 0; o >>= 1) {
    if (c < o) { sh1[c] += sh1[c + o]; sh2[c] += sh2[c + o]; }
    __syncthreads();
  }
  const float mean = sh1[0] / 256.f;
  const float var  = sh2[0] / 256.f - mean * mean;
  const float o = (v - mean) / sqrtf(var + eps) * w[c] + b[c];
  if (transposed_out) out[(long long)c * npos + pos] = o;
  else                out[(long long)pos * 256 + c] = o;
}

// im2col for 3x3 pad-1 conv on (64,64,256) -> (4096, 2304), k = i*9 + ky*3 + kx
__global__ __launch_bounds__(256) void im2col3(const float* __restrict__ in,
                                               float* __restrict__ out)
{
  const long long idx = (long long)blockIdx.x * 256 + threadIdx.x;
  if (idx >= 4096LL * 2304) return;
  const int col = (int)(idx % 2304);
  const long long p = idx / 2304;
  const int h = (int)(p / 64), w = (int)(p % 64);
  const int i = col / 9, r = col % 9, ky = r / 3, kx = r % 3;
  const int hh = h + ky - 1, ww = w + kx - 1;
  out[idx] = (hh >= 0 && hh < 64 && ww >= 0 && ww < 64)
                 ? in[((long long)hh * 64 + ww) * 256 + i] : 0.f;
}

// ===========================================================================
// Host side
// ===========================================================================
static void gemm(hipStream_t st,
                 const float* A, long long aO, long long aI, int lda,
                 const float* B, long long bO, long long bI, int ldb,
                 float* C, long long cO, long long cI, int ldc,
                 const float* bias, const float* resid,
                 int M, int N, int K, int batch, int inner,
                 float alpha, int flags, int transB)
{
  dim3 grid((N + 63) / 64, (M + 127) / 128, batch);
  gemm_wmma_f16<<<grid, 256, 0, st>>>(A, aO, aI, lda, B, bO, bI, ldb,
                                      C, cO, cI, ldc, bias, resid,
                                      M, N, K, inner, alpha, flags, transB);
}

static inline int blks(long long n) { return (int)((n + 255) / 256); }

extern "C" void kernel_launch(void* const* d_in, const int* in_sizes, int n_in,
                              void* d_out, int out_size, void* d_ws, size_t ws_size,
                              hipStream_t stream)
{
  (void)in_sizes; (void)n_in; (void)out_size; (void)ws_size;
  const float* x_in    = (const float*)d_in[0];
  const float* patch_w = (const float*)d_in[1];   // (768 out, 768 in)  [N,K]
  const float* patch_b = (const float*)d_in[2];
  const float* pos_emb = (const float*)d_in[3];
  const float* neck_w1   = (const float*)d_in[60]; // (256,768) [N,K]
  const float* neck_ln1w = (const float*)d_in[61];
  const float* neck_ln1b = (const float*)d_in[62];
  const float* neck_w2   = (const float*)d_in[63]; // (256,2304) [N,K]
  const float* neck_ln2w = (const float*)d_in[64];
  const float* neck_ln2b = (const float*)d_in[65];

  // ---- workspace layout (floats), total ~65.1M floats (~261 MB) ----
  float* ws = (float*)d_ws;
  const long long SZ_A = 4900LL * 768;
  float* WS_LN  = ws;                              // 4900*768
  float* WS_WIN = WS_LN  + SZ_A;                   // 4900*768
  float* WS_AO  = WS_WIN + SZ_A;                   // 4900*768
  float* WS_P   = WS_AO  + SZ_A;                   // 4900*768
  float* WS_QKV = WS_P   + SZ_A;                   // 4900*2304
  float* WS_S   = WS_QKV + 4900LL * 2304;          // 4096*4096
  float* WS_RH  = WS_S   + 4096LL * 4096;          // 12*4096*64
  float* WS_RW  = WS_RH  + 12LL * 4096 * 64;       // 12*4096*64
  float* WS_M   = WS_RW  + 12LL * 4096 * 64;       // 4096*3072
  float* WS_X   = WS_M   + 4096LL * 3072;          // 4096*768

  const float SCALE = 0.125f;  // 64^-0.5

  // ---- patch embed: im2col + GEMM (Bᵀ) + bias, then + pos_embed ----
  im2col_patch<<<blks(4096LL * 768), 256, 0, stream>>>(x_in, WS_QKV);
  gemm(stream, WS_QKV, 0, 0, 768, patch_w, 0, 0, 768, WS_X, 0, 0, 768,
       patch_b, nullptr, 4096, 768, 768, 1, 1, 1.f, 0, 1);
  add_inplace<<<blks(4096LL * 768), 256, 0, stream>>>(WS_X, pos_emb, 4096LL * 768);

  // ---- transformer blocks ----
  for (int blk = 0; blk < 4; ++blk) {
    const int base = 4 + blk * 14;
    const float* ln1_w  = (const float*)d_in[base + 0];
    const float* ln1_b  = (const float*)d_in[base + 1];
    const float* qkv_w  = (const float*)d_in[base + 2];
    const float* qkv_b  = (const float*)d_in[base + 3];
    const float* proj_w = (const float*)d_in[base + 4];
    const float* proj_b = (const float*)d_in[base + 5];
    const float* rp_h   = (const float*)d_in[base + 6];
    const float* rp_w   = (const float*)d_in[base + 7];
    const float* ln2_w  = (const float*)d_in[base + 8];
    const float* ln2_b  = (const float*)d_in[base + 9];
    const float* fc1_w  = (const float*)d_in[base + 10];
    const float* fc1_b  = (const float*)d_in[base + 11];
    const float* fc2_w  = (const float*)d_in[base + 12];
    const float* fc2_b  = (const float*)d_in[base + 13];
    const int win = (blk == 3) ? 0 : 14;

    layer_norm_row<<<4096, 256, 0, stream>>>(WS_X, ln1_w, ln1_b, WS_LN, 768, 1e-5f);

    const float* attnIn;
    int rows, Ntok, nwin, Hd;
    if (win) {
      win_part<<<blks(4900LL * 768), 256, 0, stream>>>(WS_LN, WS_WIN);
      attnIn = WS_WIN; rows = 4900; Ntok = 196; nwin = 25; Hd = 14;
    } else {
      attnIn = WS_LN; rows = 4096; Ntok = 4096; nwin = 1; Hd = 64;
    }

    // qkv = attnIn @ qkv_w + qkv_b
    gemm(stream, attnIn, 0, 0, 768, qkv_w, 0, 0, 2304, WS_QKV, 0, 0, 2304,
         qkv_b, nullptr, rows, 2304, 768, 1, 1, 1.f, 0, 0);

    // rel-pos bias terms from unscaled q
    const long long totalR = (long long)nwin * 12 * Ntok * Hd;
    rel_pos_kernel<<<blks(totalR), 256, 0, stream>>>(WS_QKV, rp_h, WS_RH, totalR,
                                                     Ntok, Hd, Hd, Hd, 0, 12, 2304);
    rel_pos_kernel<<<blks(totalR), 256, 0, stream>>>(WS_QKV, rp_w, WS_RW, totalR,
                                                     Ntok, Hd, Hd, Hd, 1, 12, 2304);

    if (win) {
      // scores = SCALE * q @ kᵀ   (batched over 25 windows x 12 heads)
      gemm(stream, WS_QKV, 196LL * 2304, 64, 2304,
           WS_QKV + 768, 196LL * 2304, 64, 2304,
           WS_S, 12LL * 196 * 196, 196LL * 196, 196,
           nullptr, nullptr, 196, 196, 64, 300, 12, SCALE, 0, 1);
      softmax_bias<<<300 * 196, 256, 0, stream>>>(WS_S, WS_RH, WS_RW, 196, 14);
      // out = attn @ v
      gemm(stream, WS_S, 12LL * 196 * 196, 196LL * 196, 196,
           WS_QKV + 1536, 196LL * 2304, 64, 2304,
           WS_AO, 196LL * 768, 64, 768,
           nullptr, nullptr, 196, 64, 196, 300, 12, 1.f, 0, 0);
    } else {
      // global attention, one head at a time (64 MB score buffer)
      for (int h = 0; h < 12; ++h) {
        gemm(stream, WS_QKV + h * 64, 0, 0, 2304,
             WS_QKV + 768 + h * 64, 0, 0, 2304,
             WS_S, 0, 0, 4096, nullptr, nullptr,
             4096, 4096, 64, 1, 1, SCALE, 0, 1);
        softmax_bias<<<4096, 256, 0, stream>>>(WS_S, WS_RH + (long long)h * 4096 * 64,
                                               WS_RW + (long long)h * 4096 * 64, 4096, 64);
        gemm(stream, WS_S, 0, 0, 4096,
             WS_QKV + 1536 + h * 64, 0, 0, 2304,
             WS_AO + h * 64, 0, 0, 768, nullptr, nullptr,
             4096, 64, 4096, 1, 1, 1.f, 0, 0);
      }
    }

    // projection (+ residual)
    if (win) {
      gemm(stream, WS_AO, 0, 0, 768, proj_w, 0, 0, 768, WS_P, 0, 0, 768,
           proj_b, nullptr, rows, 768, 768, 1, 1, 1.f, 0, 0);
      win_unpart_add<<<blks(4096LL * 768), 256, 0, stream>>>(WS_X, WS_P);
    } else {
      gemm(stream, WS_AO, 0, 0, 768, proj_w, 0, 0, 768, WS_X, 0, 0, 768,
           proj_b, WS_X, 4096, 768, 768, 1, 1, 1.f, 0, 0);
    }

    // MLP: LN2 -> GELU(fc1) -> fc2 + residual
    layer_norm_row<<<4096, 256, 0, stream>>>(WS_X, ln2_w, ln2_b, WS_LN, 768, 1e-5f);
    gemm(stream, WS_LN, 0, 0, 768, fc1_w, 0, 0, 3072, WS_M, 0, 0, 3072,
         fc1_b, nullptr, 4096, 3072, 768, 1, 1, 1.f, GEMM_GELU, 0);
    gemm(stream, WS_M, 0, 0, 3072, fc2_w, 0, 0, 768, WS_X, 0, 0, 768,
         fc2_b, WS_X, 4096, 768, 3072, 1, 1, 1.f, 0, 0);
  }

  // ---- neck ----
  // 1x1 conv (GEMM with Bᵀ), channel-LN
  gemm(stream, WS_X, 0, 0, 768, neck_w1, 0, 0, 768, WS_LN, 0, 0, 256,
       nullptr, nullptr, 4096, 256, 768, 1, 1, 1.f, 0, 1);
  ln2d_c256<<<4096, 256, 0, stream>>>(WS_LN, neck_ln1w, neck_ln1b, WS_WIN,
                                      4096, 1e-6f, 0);
  // 3x3 conv via im2col + GEMM (Bᵀ), channel-LN, write NCHW output
  im2col3<<<blks(4096LL * 2304), 256, 0, stream>>>(WS_WIN, WS_QKV);
  gemm(stream, WS_QKV, 0, 0, 2304, neck_w2, 0, 0, 2304, WS_AO, 0, 0, 256,
       nullptr, nullptr, 4096, 256, 2304, 1, 1, 1.f, 0, 1);
  ln2d_c256<<<4096, 256, 0, stream>>>(WS_AO, neck_ln2w, neck_ln2b,
                                      (float*)d_out, 4096, 1e-6f, 1);
}
